// Attention2D_50989851738214
// MI455X (gfx1250) — matrix operand order repaired and checked by reading, not compiled, not run on hardware
//
#include <hip/hip_runtime.h>
#include <hip/hip_bf16.h>

typedef __bf16 bf16;
typedef __attribute__((ext_vector_type(16))) __bf16 v16bf;
typedef __attribute__((ext_vector_type(8)))  __bf16 v8bf;
typedef __attribute__((ext_vector_type(8)))  float  v8f;
typedef __attribute__((ext_vector_type(4)))  int    v4i;

#define BATCH 8
#define DIM_  256
#define NPIX  1024
#define NH    8
#define DH    32
#define SCALE_F 0.17677669529663687f  // (256/8)^-0.5 = 32^-0.5

#if defined(__has_builtin)
#if __has_builtin(__builtin_amdgcn_global_load_async_to_lds_b128) && \
    __has_builtin(__builtin_amdgcn_s_wait_asynccnt)
#define HAVE_ASYNC 1
#endif
#endif
#ifndef HAVE_ASYNC
#define HAVE_ASYNC 0
#endif

#define AS1 __attribute__((address_space(1)))
#define AS3 __attribute__((address_space(3)))

// ---------------- WMMA fragment helpers (wave32, bf16 16x16x32) -------------

// A fragment: 16x32 (MxK), row-major src, leading dim ld; p0 -> (m0,k0).
// Lane m=lane&15; low lanes K={0..7,16..23}, high lanes K={8..15,24..31}.
__device__ inline v16bf load_frag_a(const bf16* p0, int ld) {
    int lane = threadIdx.x & 31;
    int m  = lane & 15;
    int kb = (lane < 16) ? 0 : 8;
    const bf16* p = p0 + m * ld + kb;
    union { v16bf v; v8bf h[2]; } u;
    u.h[0] = *(const v8bf*)(p);        // K = kb .. kb+7
    u.h[1] = *(const v8bf*)(p + 16);   // K = kb+16 .. kb+23
    return u.v;
}

// B fragment: 32x16 (KxN); src stored transposed (rows of N x K, K-contig),
// leading dim ld; p0 -> (n0,k0). Lane n=lane&15; low lanes K={0..7,16..23}, high K={8..15,24..31}.
__device__ inline v16bf load_frag_b(const bf16* p0, int ld) {
    int lane = threadIdx.x & 31;
    int n  = lane & 15;
    int kb = (lane < 16) ? 0 : 8;
    const bf16* p = p0 + n * ld + kb;
    union { v16bf v; v8bf h[2]; } u;
    u.h[0] = *(const v8bf*)(p);        // K = kb .. kb+7
    u.h[1] = *(const v8bf*)(p + 16);   // K = kb+16 .. kb+23
    return u.v;
}

__device__ inline v8f wmma_bf16(v16bf a, v16bf b, v8f c) {
    return __builtin_amdgcn_wmma_f32_16x16x32_bf16(
        false, a, false, b, (short)0, c, false, false);
}

// ---------------- Kernel 1: convert weights to bf16 ------------------------

__global__ void k_convert_w(const float* __restrict__ wqkv,
                            const float* __restrict__ wout,
                            bf16* __restrict__ wqkv_b,
                            bf16* __restrict__ wout_b) {
    int i = blockIdx.x * blockDim.x + threadIdx.x;
    if (i < 768 * 256) wqkv_b[i] = (bf16)wqkv[i];
    if (i < 256 * 256) wout_b[i] = (bf16)wout[i];
}

// ---------------- Kernel 2: transpose x[b][c][n] -> xb[b][n][c] (bf16) -----

__global__ void k_transpose_x(const float* __restrict__ x, bf16* __restrict__ xb) {
    __shared__ float t[32][33];
    int b  = blockIdx.z;
    int c0 = blockIdx.y * 32;
    int n0 = blockIdx.x * 32;
    int tx = threadIdx.x, ty = threadIdx.y;
    t[ty][tx] = x[((size_t)b * DIM_ + (c0 + ty)) * NPIX + n0 + tx];
    __syncthreads();
    xb[((size_t)b * NPIX + (n0 + ty)) * DIM_ + c0 + tx] = (bf16)t[tx][ty];
}

// ---------------- Kernel 3: QKV projection GEMM -----------------------------
// M=768, K=256, N=1024 per batch. Each wave: 16x64 strip (A reused x4).
// Scatter: q->qT[b][h][n][d]*SCALE, k->kT[b][h][n][d], v->v[b][h][d][n].

__global__ __launch_bounds__(256) void k_qkv(const bf16* __restrict__ wqkv_b,
                                             const bf16* __restrict__ xb,
                                             bf16* __restrict__ qT,
                                             bf16* __restrict__ kT,
                                             bf16* __restrict__ vv) {
    int wid = blockIdx.x * (blockDim.x >> 5) + (threadIdx.x >> 5);
    int b   = wid / (48 * 16);
    int rem = wid % (48 * 16);
    int m0  = (rem >> 4) * 16;     // 48 m-tiles
    int n0  = (rem & 15) * 64;     // 16 groups of 64 cols
    const bf16* A  = wqkv_b + (size_t)m0 * DIM_;
    const bf16* Bm = xb + (size_t)b * NPIX * DIM_ + (size_t)n0 * DIM_;
    v8f acc[4] = {};
#pragma unroll
    for (int k = 0; k < DIM_; k += 32) {
        v16bf a = load_frag_a(A + k, DIM_);
#pragma unroll
        for (int t = 0; t < 4; ++t)
            acc[t] = wmma_bf16(a, load_frag_b(Bm + (size_t)t * 16 * DIM_ + k, DIM_), acc[t]);
    }
    int lane = threadIdx.x & 31;
    int col  = lane & 15;
    int mo   = (lane < 16) ? 0 : 8;
#pragma unroll
    for (int t = 0; t < 4; ++t) {
        int n = n0 + t * 16 + col;
#pragma unroll
        for (int r = 0; r < 8; ++r) {
            int o = m0 + mo + r;
            float val = acc[t][r];
            if (o < 256) {
                int h = o >> 5, d = o & 31;
                qT[(((size_t)b * NH + h) * NPIX + n) * DH + d] = (bf16)(val * SCALE_F);
            } else if (o < 512) {
                int oo = o - 256; int h = oo >> 5, d = oo & 31;
                kT[(((size_t)b * NH + h) * NPIX + n) * DH + d] = (bf16)val;
            } else {
                int oo = o - 512; int h = oo >> 5, d = oo & 31;
                vv[(((size_t)b * NH + h) * DH + d) * NPIX + n] = (bf16)val;
            }
        }
    }
}

// ---------------- Kernel 4: fused flash attention ---------------------------
// Block = 4 waves = rows i0..i0+63 of one (b,h). K/V 32x32 tiles staged to LDS
// cooperatively (async global->LDS, double-buffered), shared by all 4 waves.
// Per wave: online softmax + P staged to LDS as bf16 A-fragment.

__device__ inline void stage_tile(const bf16* __restrict__ Kp,
                                  const bf16* __restrict__ Vp,
                                  bf16* kb, bf16* vb, int j0, int tid) {
    int t    = tid & 63;
    int row  = t >> 1;
    int half = t & 1;
    const bf16* src = (tid < 64) ? (Kp + (size_t)(j0 + row) * DH + half * 16)
                                 : (Vp + (size_t)row * NPIX + j0 + half * 16);
    bf16* dst = ((tid < 64) ? kb : vb) + row * 32 + half * 16;
#if HAVE_ASYNC
    __builtin_amdgcn_global_load_async_to_lds_b128((AS1 v4i*)src, (AS3 v4i*)dst, 0, 0);
#else
    *(v8bf*)dst = *(const v8bf*)src;
#endif
}

__device__ inline void wait_stage() {
#if HAVE_ASYNC
    __builtin_amdgcn_s_wait_asynccnt(0);
#endif
}

__global__ __launch_bounds__(128) void k_attn(const bf16* __restrict__ qT,
                                              const bf16* __restrict__ kT,
                                              const bf16* __restrict__ vv,
                                              bf16* __restrict__ ao) {
    __shared__ __align__(16) bf16 kbuf[2][32 * 32];
    __shared__ __align__(16) bf16 vbuf[2][32 * 32];
    __shared__ __align__(16) bf16 pst[4][16 * 32];
    int tid  = threadIdx.x;
    int wave = tid >> 5;
    int lane = tid & 31;
    int bh   = blockIdx.x >> 4;          // 16 blocks per (b,h)
    int blk  = blockIdx.x & 15;
    int b = bh >> 3, h = bh & 7;
    int i0 = (blk * 4 + wave) * 16;

    const bf16* Q = qT + ((size_t)b * NH + h) * NPIX * DH;
    const bf16* K = kT + ((size_t)b * NH + h) * NPIX * DH;
    const bf16* V = vv + ((size_t)b * NH + h) * DH * NPIX;

    v16bf aq = load_frag_a(Q + (size_t)i0 * DH, DH);   // 16x32, SCALE folded in
    v8f acc0 = {}, acc1 = {};
    float mrow[8], lrow[8];
#pragma unroll
    for (int r = 0; r < 8; ++r) { mrow[r] = -1e30f; lrow[r] = 0.0f; }

    bf16* stage = pst[wave];
    int col = lane & 15;
    int rhi = (lane < 16) ? 0 : 8;
    v8f zero = {};

    stage_tile(K, V, kbuf[0], vbuf[0], 0, tid);        // prologue: tile 0

    for (int jt = 0; jt < NPIX / 32; ++jt) {
        int cur = jt & 1;
        wait_stage();          // tile jt landed (this wave's async op)
        __syncthreads();       // all waves' data visible; prev reads of buf done
        if (jt + 1 < NPIX / 32)
            stage_tile(K, V, kbuf[1 - cur], vbuf[1 - cur], (jt + 1) * 32, tid);

        const bf16* kb = kbuf[cur];
        const bf16* vb = vbuf[cur];
        v16bf bk0 = load_frag_b(kb, DH);             // j rows 0..15 (local)
        v16bf bk1 = load_frag_b(kb + 16 * DH, DH);   // j rows 16..31
        v8f s0 = wmma_bf16(aq, bk0, zero);
        v8f s1 = wmma_bf16(aq, bk1, zero);
#pragma unroll
        for (int r = 0; r < 8; ++r) {
            float a0 = s0[r], a1 = s1[r];
            float tmax = fmaxf(a0, a1);
#pragma unroll
            for (int off = 1; off < 16; off <<= 1)
                tmax = fmaxf(tmax, __shfl_xor(tmax, off, 16));
            float newm = fmaxf(mrow[r], tmax);
            float corr = __expf(mrow[r] - newm);
            float p0 = __expf(a0 - newm);
            float p1 = __expf(a1 - newm);
            float rs = p0 + p1;
#pragma unroll
            for (int off = 1; off < 16; off <<= 1)
                rs += __shfl_xor(rs, off, 16);
            lrow[r] = lrow[r] * corr + rs;
            mrow[r] = newm;
            acc0[r] *= corr;
            acc1[r] *= corr;
            int row = r + rhi;
            stage[row * 32 + col]      = (bf16)p0;
            stage[row * 32 + 16 + col] = (bf16)p1;
        }
        asm volatile("s_wait_dscnt 0" ::: "memory");   // cross-lane LDS handoff
        v16bf ap  = load_frag_a(stage, 32);            // P tile 16x32 (A frag)
        v16bf bv0 = load_frag_b(vb, 32);               // d = 0..15
        v16bf bv1 = load_frag_b(vb + 16 * 32, 32);     // d = 16..31
        acc0 = wmma_bf16(ap, bv0, acc0);
        acc1 = wmma_bf16(ap, bv1, acc1);
    }

#pragma unroll
    for (int r = 0; r < 8; ++r) {
        float inv = 1.0f / lrow[r];
        int row = i0 + r + rhi;
        size_t base = ((size_t)b * NPIX + row) * DIM_ + (size_t)h * DH;
        ao[base + col]      = (bf16)(acc0[r] * inv);
        ao[base + 16 + col] = (bf16)(acc1[r] * inv);
    }
}

// ---------------- Kernel 5: output projection + bias ------------------------
// y[b,o,n] = sum_c2 wout[o,c2] * ao[b,n,c2] + bout[o]; M=256, K=256, N=1024.
// Each wave: 16x64 strip.

__global__ __launch_bounds__(256) void k_proj(const bf16* __restrict__ wout_b,
                                              const bf16* __restrict__ ao,
                                              const float* __restrict__ bout,
                                              float* __restrict__ y) {
    int wid = blockIdx.x * (blockDim.x >> 5) + (threadIdx.x >> 5);
    int b   = wid / (16 * 16);
    int rem = wid % (16 * 16);
    int m0  = (rem >> 4) * 16;
    int n0  = (rem & 15) * 64;
    const bf16* A  = wout_b + (size_t)m0 * DIM_;
    const bf16* Bm = ao + (size_t)b * NPIX * DIM_ + (size_t)n0 * DIM_;
    v8f acc[4] = {};
#pragma unroll
    for (int k = 0; k < DIM_; k += 32) {
        v16bf a = load_frag_a(A + k, DIM_);
#pragma unroll
        for (int t = 0; t < 4; ++t)
            acc[t] = wmma_bf16(a, load_frag_b(Bm + (size_t)t * 16 * DIM_ + k, DIM_), acc[t]);
    }
    int lane = threadIdx.x & 31;
    int col  = lane & 15;
    int mo   = (lane < 16) ? 0 : 8;
#pragma unroll
    for (int t = 0; t < 4; ++t) {
        int n = n0 + t * 16 + col;
#pragma unroll
        for (int r = 0; r < 8; ++r) {
            int o = m0 + mo + r;
            y[((size_t)b * DIM_ + o) * NPIX + n] = acc[t][r] + bout[o];
        }
    }
}

// ---------------- Launcher ---------------------------------------------------

extern "C" void kernel_launch(void* const* d_in, const int* in_sizes, int n_in,
                              void* d_out, int out_size, void* d_ws, size_t ws_size,
                              hipStream_t stream) {
    const float* x    = (const float*)d_in[0];
    const float* wqkv = (const float*)d_in[1];
    const float* wout = (const float*)d_in[2];
    const float* bout = (const float*)d_in[3];
    float* y = (float*)d_out;

    char* ws = (char*)d_ws;
    bf16* wqkv_b = (bf16*)(ws + 0);          // 384 KB
    bf16* wout_b = (bf16*)(ws + 393216);     // 128 KB
    bf16* xb     = (bf16*)(ws + 524288);     // 4 MB
    bf16* qT     = (bf16*)(ws + 4718592);    // 4 MB
    bf16* kT     = (bf16*)(ws + 8912896);    // 4 MB
    bf16* vv     = (bf16*)(ws + 13107200);   // 4 MB
    bf16* ao     = (bf16*)(ws + 17301504);   // 4 MB (total ~20.5 MB)

    k_convert_w<<<768, 256, 0, stream>>>(wqkv, wout, wqkv_b, wout_b);
    k_transpose_x<<<dim3(32, 8, 8), dim3(32, 32), 0, stream>>>(x, xb);
    k_qkv<<<768, 256, 0, stream>>>(wqkv_b, xb, qT, kT, vv);
    k_attn<<<1024, 128, 0, stream>>>(qT, kT, vv, ao);
    k_proj<<<256, 256, 0, stream>>>(wout_b, ao, bout, y);
}
